// AnExNet_89378269429994
// MI455X (gfx1250) — compile-verified, matrix operand order checked by reading, not run on hardware
//
#include <hip/hip_runtime.h>
#include <hip/hip_bf16.h>
#include <stdint.h>

// ---------------- types ----------------
typedef __attribute__((ext_vector_type(16))) __bf16       v16bf;
typedef __attribute__((ext_vector_type(8)))  float        v8f;
typedef __attribute__((ext_vector_type(8)))  unsigned int v8u;

__device__ __forceinline__ unsigned short f2bf(float f) {
  unsigned int u = __float_as_uint(f);
  u += 0x7FFFu + ((u >> 16) & 1u);
  return (unsigned short)(u >> 16);
}
__device__ __forceinline__ float sigm(float x) { return 1.f / (1.f + expf(-x)); }

// ---------------- weight packing into WMMA A-register image ----------------
// co-tile-major layout so each wave's A stream is contiguous:
//   wp[((ct * taps*kChunks) + (tap*kChunks + kc)) * 256 + lane*8 + r]
// lane m = lane&15 (row M), half = lane>>4 ; VGPR r holds packed K pair:
//   k = (r>=4 ? 16 : 0) + half*8 + 2*(r&3)
__global__ void pack_conv_w(const float* __restrict__ w, unsigned int* __restrict__ wp,
                            int Cout, int CinTotal, int ciOff, int ciCnt, int ks) {
  int taps = ks * ks;
  int coTiles = (Cout + 15) >> 4;
  int kChunks = (ciCnt + 31) >> 5;
  int ikTotal = taps * kChunks;
  long long total = (long long)coTiles * ikTotal * 256;
  for (long long idx = (long long)blockIdx.x * blockDim.x + threadIdx.x; idx < total;
       idx += (long long)gridDim.x * blockDim.x) {
    int r    = (int)(idx & 7);
    int lane = (int)((idx >> 3) & 31);
    long long rest = idx >> 8;
    int it = (int)(rest % ikTotal);
    int ct = (int)(rest / ikTotal);
    int kc = it % kChunks;
    int tap = it / kChunks;
    int m = lane & 15, half = lane >> 4;
    int k = ((r >= 4) ? 16 : 0) + half * 8 + 2 * (r & 3);
    int co = ct * 16 + m;
    unsigned int pk = 0;
    for (int t = 0; t < 2; ++t) {
      int ci = kc * 32 + k + t;
      float val = 0.f;
      if (co < Cout && ci < ciCnt)
        val = w[((long long)co * CinTotal + (ciOff + ci)) * taps + tap];
      pk |= ((unsigned int)f2bf(val)) << (16 * t);
    }
    wp[idx] = pk;
  }
}

// ---------------- implicit-GEMM conv via WMMA bf16 ----------------
// 256 threads = 8 waves; wave owns 2 co-tiles (up to 256 out ch per block).
// B fragment (16 pixels x 32 k, bf16 activations) double-buffered in LDS.
// A addressing: contiguous stream per co-tile, pointer += 256 dwords / iter.
__global__ __launch_bounds__(256) void conv2d_wmma(
    const unsigned short* __restrict__ x, const unsigned int* __restrict__ wp,
    const float* __restrict__ bias, float* __restrict__ outF,
    unsigned short* __restrict__ outB,
    int Nimg, int Cin, int H, int W, int Cout, int Ho, int Wo,
    int ks, int stride, int pad, int relu, int accumulate) {
  __shared__ unsigned int ldsB[2][256];
  int HWo = Ho * Wo;
  int pixTiles = (HWo + 15) >> 4;
  int coTiles  = (Cout + 15) >> 4;
  int coGroups = (coTiles + 15) >> 4;
  int bid = blockIdx.x;
  int cg = bid % coGroups; bid /= coGroups;
  int pt = bid % pixTiles;
  int n  = bid / pixTiles;
  int tid = threadIdx.x;
  int wave = tid >> 5, lane = tid & 31;
  int ct0 = cg * 16 + wave;
  int ct1 = cg * 16 + 8 + wave;
  bool a0 = ct0 < coTiles, a1 = ct1 < coTiles;
  int kChunks = (Cin + 31) >> 5;
  int total = ks * ks * kChunks;
  long long HW = (long long)H * W;
  const unsigned short* xn = x + (long long)n * Cin * HW;

  long long ctStride = (long long)total * 256;
  const unsigned int* ap0 = wp + (long long)ct0 * ctStride + lane * 8;
  const unsigned int* ap1 = wp + (long long)ct1 * ctStride + lane * 8;

  // per-thread B-build constants: this thread builds LDS dword d = tid
  int d = tid & 255;
  int rB = d & 7, lnB = d >> 3;
  int colB = lnB & 15, halfB = lnB >> 4;
  int kB = ((rB >= 4) ? 16 : 0) + halfB * 8 + 2 * (rB & 3);
  int pB = pt * 16 + colB;
  bool pValid = pB < HWo;
  int hoB = pValid ? pB / Wo : 0;
  int woB = pValid ? pB % Wo : 0;
  int bh0 = hoB * stride - pad;   // ih = bh0 + kh
  int bw0 = woB * stride - pad;   // iw = bw0 + kw
  long long kB_HW = (long long)kB * HW;
  long long HW32  = 32LL * HW;

  // incrementally-advanced iteration state (kc fastest, then kw, then kh)
  int kc32 = 0, khS = 0, kwS = 0;
  long long kcOff = 0;
  int kcLim = kChunks * 32;
  auto advance = [&]() {
    kc32 += 32; kcOff += HW32;
    if (kc32 == kcLim) {
      kc32 = 0; kcOff = 0;
      if (++kwS == ks) { kwS = 0; ++khS; }
    }
  };
  auto build = [&](int buf) {
    unsigned int pk = 0;
    if (pValid) {
      int ih = bh0 + khS, iw = bw0 + kwS;
      if ((unsigned)ih < (unsigned)H && (unsigned)iw < (unsigned)W) {
        int ci0 = kc32 + kB;
        long long addr = kcOff + kB_HW + (long long)ih * W + iw;
        unsigned int v0 = (ci0 < Cin) ? (unsigned int)xn[addr] : 0u;
        unsigned int v1 = (ci0 + 1 < Cin) ? (unsigned int)xn[addr + HW] : 0u;
        pk = v0 | (v1 << 16);
      }
    }
    ldsB[buf][d] = pk;
  };

  v8f acc0 = {}, acc1 = {};
  build(0); advance();
  __syncthreads();
  for (int it = 0; it < total; ++it) {
    v8u breg;
#pragma unroll
    for (int q = 0; q < 8; ++q) breg[q] = ldsB[it & 1][lane * 8 + q];
    v16bf b = __builtin_bit_cast(v16bf, breg);
    if (a0) {
      __builtin_prefetch(ap0 + 256, 0, 1);
      v8u areg;
#pragma unroll
      for (int q = 0; q < 8; ++q) areg[q] = ap0[q];
      acc0 = __builtin_amdgcn_wmma_f32_16x16x32_bf16(
          false, __builtin_bit_cast(v16bf, areg), false, b, (short)0, acc0, false, false);
    }
    if (a1) {
      v8u areg;
#pragma unroll
      for (int q = 0; q < 8; ++q) areg[q] = ap1[q];
      acc1 = __builtin_amdgcn_wmma_f32_16x16x32_bf16(
          false, __builtin_bit_cast(v16bf, areg), false, b, (short)0, acc1, false, false);
    }
    if (it + 1 < total) { build((it + 1) & 1); advance(); }
    __syncthreads();
    ap0 += 256; ap1 += 256;
  }
  // epilogue
  int col = lane & 15, half = lane >> 4;
  int p = pt * 16 + col;
  if (p < HWo) {
#pragma unroll
    for (int ti = 0; ti < 2; ++ti) {
      int ct = ti ? ct1 : ct0;
      if (ct >= coTiles) continue;
      v8f& acc = ti ? acc1 : acc0;
#pragma unroll
      for (int v = 0; v < 8; ++v) {
        int co = ct * 16 + half * 8 + v;
        if (co < Cout) {
          long long o = ((long long)n * Cout + co) * HWo + p;
          float val = acc[v];
          if (bias) val += bias[co];
          if (accumulate) val += outF[o];
          if (relu) val = fmaxf(val, 0.f);
          if (outF) outF[o] = val;
          if (outB) outB[o] = f2bf(val);
        }
      }
    }
  }
}

// ---------------- elementwise / glue kernels ----------------
__global__ void k_fill0(float* p, long long n) {
  for (long long i = (long long)blockIdx.x * blockDim.x + threadIdx.x; i < n;
       i += (long long)gridDim.x * blockDim.x) p[i] = 0.f;
}
__global__ void k_fill0h(unsigned short* p, long long n) {
  for (long long i = (long long)blockIdx.x * blockDim.x + threadIdx.x; i < n;
       i += (long long)gridDim.x * blockDim.x) p[i] = 0;
}
// out (Nimg,10,HW) bf16: ch<3 from x, ch>=3 broadcast of v (Nimg,7)
__global__ void k_concat_imgv(const float* __restrict__ x, const float* __restrict__ v,
                              unsigned short* __restrict__ out, int Nimg, int HW) {
  long long tot = (long long)Nimg * 10 * HW;
  for (long long i = (long long)blockIdx.x * blockDim.x + threadIdx.x; i < tot;
       i += (long long)gridDim.x * blockDim.x) {
    int p = (int)(i % HW);
    int ch = (int)((i / HW) % 10);
    int n = (int)(i / ((long long)HW * 10));
    float val = (ch < 3) ? x[((long long)n * 3 + ch) * HW + p] : v[n * 7 + (ch - 3)];
    out[i] = f2bf(val);
  }
}
__global__ void k_mean_hw(const float* __restrict__ in, float* __restrict__ out,
                          int NC, int HW, float scale) {
  for (int i = blockIdx.x * blockDim.x + threadIdx.x; i < NC; i += gridDim.x * blockDim.x) {
    float s = 0.f;
    const float* q = in + (long long)i * HW;
    for (int p = 0; p < HW; ++p) s += q[p];
    out[i] = s * scale;
  }
}
__global__ void k_sum_ctx(const float* __restrict__ vec, float* __restrict__ xsr,
                          int B, int Kc, int C) {
  int tot = B * C;
  for (int i = blockIdx.x * blockDim.x + threadIdx.x; i < tot; i += gridDim.x * blockDim.x) {
    int b = i / C, c = i % C;
    float s = 0.f;
    for (int k = 0; k < Kc; ++k) s += vec[((long long)(b * Kc + k)) * C + c];
    xsr[i] = s;
  }
}
__global__ void k_addfr0(const float* __restrict__ xsr, const float* __restrict__ vecT,
                         float* __restrict__ xsrf, int B, int C, int T) {
  int tot = B * C;
  for (int i = blockIdx.x * blockDim.x + threadIdx.x; i < tot; i += gridDim.x * blockDim.x) {
    int b = i / C, c = i % C;
    xsrf[i] = xsr[i] + vecT[(long long)b * T * C + c];
  }
}
__global__ void k_build_li(const float* __restrict__ xtrpre, const float* __restrict__ xsr,
                           float* __restrict__ li, int B, int T, int R) {
  long long tot = (long long)B * T * 2 * R;
  for (long long i = (long long)blockIdx.x * blockDim.x + threadIdx.x; i < tot;
       i += (long long)gridDim.x * blockDim.x) {
    int k = (int)(i % (2 * R));
    int t = (int)((i / (2 * R)) % T);
    int b = (int)(i / ((long long)(2 * R) * T));
    li[i] = (k < R) ? xtrpre[((long long)b * T + t) * R + k] : xsr[b * R + (k - R)];
  }
}
// dst bf16 [k*16+b] = (b<4) ? src[b*rowStride + srcOff + k] : 0
__global__ void k_gatherT(const float* __restrict__ src, unsigned short* __restrict__ dst,
                          int K, int rowStride, int srcOff) {
  int tot = K * 16;
  for (int i = blockIdx.x * blockDim.x + threadIdx.x; i < tot; i += gridDim.x * blockDim.x) {
    int k = i >> 4, b = i & 15;
    dst[i] = (b < 4) ? f2bf(src[(long long)b * rowStride + srcOff + k]) : (unsigned short)0;
  }
}
// gT (1024 gates x 16 cols) f32; gates i,f,g,o in quarters of 256
__global__ void k_lstm_gate(const float* __restrict__ gT, float* __restrict__ h,
                            float* __restrict__ c, float* __restrict__ ys, int t, int T) {
  int tot = 4 * 256;
  for (int i = blockIdx.x * blockDim.x + threadIdx.x; i < tot; i += gridDim.x * blockDim.x) {
    int b = i >> 8, cx = i & 255;
    float gi = gT[(0 * 256 + cx) * 16 + b];
    float gf = gT[(1 * 256 + cx) * 16 + b];
    float gg = gT[(2 * 256 + cx) * 16 + b];
    float go = gT[(3 * 256 + cx) * 16 + b];
    float cc = sigm(gf) * c[b * 256 + cx] + sigm(gi) * tanhf(gg);
    c[b * 256 + cx] = cc;
    float hh = sigm(go) * tanhf(cc);
    h[b * 256 + cx] = hh;
    ys[((long long)b * T + t) * 256 + cx] = hh;
  }
}
// repk bf16 (64,256,256): n=b*16+j -> j==0: xsrf[b], else ys2[b][j-1]; bcast HW
__global__ void k_build_repk(const float* __restrict__ xsrf, const float* __restrict__ ys2,
                             unsigned short* __restrict__ repk) {
  long long tot = 64LL * 256 * 256;
  for (long long i = (long long)blockIdx.x * blockDim.x + threadIdx.x; i < tot;
       i += (long long)gridDim.x * blockDim.x) {
    int c = (int)((i >> 8) & 255);
    int n = (int)(i >> 16);
    int b = n >> 4, j = n & 15;
    float val = (j == 0) ? xsrf[b * 256 + c] : ys2[((long long)b * 15 + (j - 1)) * 256 + c];
    repk[i] = f2bf(val);
  }
}
__global__ void k_bcast_v(const float* __restrict__ v, unsigned short* __restrict__ out,
                          int Nimg, int C, int HW) {
  long long tot = (long long)Nimg * C * HW;
  for (long long i = (long long)blockIdx.x * blockDim.x + threadIdx.x; i < tot;
       i += (long long)gridDim.x * blockDim.x) {
    int c = (int)((i / HW) % C);
    int n = (int)(i / ((long long)HW * C));
    out[i] = f2bf(v[n * C + c]);
  }
}
// ConvLSTM gates: g (N,4C,HW) f32 order i,f,g,o -> h bf16, c f32
__global__ void k_convlstm_gate(const float* __restrict__ g, unsigned short* __restrict__ h,
                                float* __restrict__ c, int Nimg, int C, int HW) {
  long long tot = (long long)Nimg * C * HW;
  for (long long i = (long long)blockIdx.x * blockDim.x + threadIdx.x; i < tot;
       i += (long long)gridDim.x * blockDim.x) {
    int p  = (int)(i % HW);
    int cc = (int)((i / HW) % C);
    int n  = (int)(i / ((long long)HW * C));
    long long base = (long long)n * 4 * C * HW;
    float gi = g[base + ((long long)(0 * C + cc)) * HW + p];
    float gf = g[base + ((long long)(1 * C + cc)) * HW + p];
    float gg = g[base + ((long long)(2 * C + cc)) * HW + p];
    float go = g[base + ((long long)(3 * C + cc)) * HW + p];
    float cn = sigm(gf) * c[i] + sigm(gi) * tanhf(gg);
    c[i] = cn;
    h[i] = f2bf(sigm(go) * tanhf(cn));
  }
}
// split (N,2Z,HW) f32 -> loc,lva f32 (outputs)
__global__ void k_split_fac(const float* __restrict__ tfac, float* __restrict__ loc,
                            float* __restrict__ lva, int Nimg, int Z, int HW) {
  long long tot = (long long)Nimg * Z * HW;
  for (long long i = (long long)blockIdx.x * blockDim.x + threadIdx.x; i < tot;
       i += (long long)gridDim.x * blockDim.x) {
    int p = (int)(i % HW);
    int c = (int)((i / HW) % Z);
    int n = (int)(i / ((long long)HW * Z));
    long long base = (long long)n * 2 * Z * HW;
    loc[i] = tfac[base + (long long)c * HW + p];
    lva[i] = tfac[base + (long long)(Z + c) * HW + p];
  }
}
// z bf16 = loc + exp(0.5*lva)*eps
__global__ void k_z(const float* __restrict__ tfac, const float* __restrict__ e,
                    unsigned short* __restrict__ z, int Nimg, int Z, int HW) {
  long long tot = (long long)Nimg * Z * HW;
  for (long long i = (long long)blockIdx.x * blockDim.x + threadIdx.x; i < tot;
       i += (long long)gridDim.x * blockDim.x) {
    int p = (int)(i % HW);
    int c = (int)((i / HW) % Z);
    int n = (int)(i / ((long long)HW * Z));
    long long base = (long long)n * 2 * Z * HW;
    float loc = tfac[base + (long long)c * HW + p];
    float lva = tfac[base + (long long)(Z + c) * HW + p];
    z[i] = f2bf(loc + expf(0.5f * lva) * e[i]);
  }
}

// ---------------- host orchestration ----------------
extern "C" void kernel_launch(void* const* d_in, const int* in_sizes, int n_in,
                              void* d_out, int out_size, void* d_ws, size_t ws_size,
                              hipStream_t stream) {
  (void)in_sizes; (void)n_in; (void)out_size; (void)ws_size;
  const float* xsk = (const float*)d_in[3];
  const float* vsk = (const float*)d_in[4];
  const float* xtk = (const float*)d_in[5];
  const float* vtk = (const float*)d_in[6];
  const float* xq  = (const float*)d_in[7];
  const float* vq  = (const float*)d_in[8];
  const float* eps = (const float*)d_in[9];
  const float* P[44];
  for (int j = 0; j < 44; ++j) P[j] = (const float*)d_in[10 + j];
  enum {
    i_se_w1 = 0, i_se_b1, i_se_w2, i_se_b2, i_se_w3, i_se_b3,
    i_in_w1, i_in_b1, i_in_w2, i_in_b2,
    i_l0_wih, i_l0_whh, i_l0_bih, i_l0_bhh,
    i_l1_wih, i_l1_whh, i_l1_bih, i_l1_bhh,
    i_pf_w1, i_pf_b1, i_pf_w2, i_pf_b2, i_pf_w3, i_pf_b3,
    i_qf_w1, i_qf_b1, i_qf_w2, i_qf_b2, i_qf_w3, i_qf_b3,
    i_ic_wx, i_ic_wh, i_ic_b, i_gc_wx, i_gc_wh, i_gc_b,
    i_gd_w, i_gd_b, i_go_w1, i_go_b1, i_go_w2, i_go_b2, i_go_w3, i_go_b3
  };
  const int NQ = 64, Tt = 15, Kc = 5, Bb = 4, AR = 16;

  // deterministic linear workspace allocator
  char* cur = (char*)d_ws;
  auto allocF = [&](long long n) -> float* {
    float* r = (float*)cur; cur += ((n * 4 + 255) / 256) * 256; return r;
  };
  auto allocH = [&](long long n) -> unsigned short* {
    unsigned short* r = (unsigned short*)cur; cur += ((n * 2 + 255) / 256) * 256; return r;
  };
  auto allocU = [&](long long n) -> unsigned int* {
    unsigned int* r = (unsigned int*)cur; cur += ((n * 4 + 255) / 256) * 256; return r;
  };
  auto packDW = [](int Cout, int ci, int ks) -> long long {
    return (long long)ks * ks * ((Cout + 15) / 16) * ((ci + 31) / 32) * 256;
  };
  auto packW = [&](const float* w, int Cout, int CinTot, int ciOff, int ciCnt,
                   int ks) -> const unsigned int* {
    long long n = packDW(Cout, ciCnt, ks);
    unsigned int* dst = allocU(n);
    int blocks = (int)((n + 255) / 256); if (blocks > 2048) blocks = 2048;
    pack_conv_w<<<blocks, 256, 0, stream>>>(w, dst, Cout, CinTot, ciOff, ciCnt, ks);
    return dst;
  };
  auto conv = [&](const unsigned short* x, const unsigned int* wp, const float* bias,
                  float* outF, unsigned short* outB,
                  int Nimg, int Cin, int H, int W, int Cout, int Ho, int Wo,
                  int ks, int stride, int pad, int relu, int acc) {
    int pixTiles = (Ho * Wo + 15) / 16;
    int coTiles = (Cout + 15) / 16;
    int coGroups = (coTiles + 15) / 16;
    unsigned blocks = (unsigned)((long long)Nimg * pixTiles * coGroups);
    conv2d_wmma<<<blocks, 256, 0, stream>>>(x, wp, bias, outF, outB, Nimg, Cin, H, W,
                                            Cout, Ho, Wo, ks, stride, pad, relu, acc);
  };
  auto g256 = [](long long n) { unsigned b = (unsigned)((n + 255) / 256); return b > 8192u ? 8192u : b; };
  auto fill0 = [&](float* p, long long n) { k_fill0<<<g256(n), 256, 0, stream>>>(p, n); };
  auto fill0h = [&](unsigned short* p, long long n) { k_fill0h<<<g256(n), 256, 0, stream>>>(p, n); };

  // ---- packed weights (bf16 WMMA register image, co-tile-major) ----
  const unsigned int* pk_se1 = packW(P[i_se_w1], 64, 10, 0, 10, 5);
  const unsigned int* pk_se2 = packW(P[i_se_w2], 64, 64, 0, 64, 5);
  const unsigned int* pk_se3 = packW(P[i_se_w3], 256, 64, 0, 64, 5);
  const unsigned int* pk_in1 = packW(P[i_in_w1], 64, 10, 0, 10, 5);
  const unsigned int* pk_in2 = packW(P[i_in_w2], 256, 64, 0, 64, 5);
  const unsigned int* pk_pf1 = packW(P[i_pf_w1], 128, 64, 0, 64, 5);
  const unsigned int* pk_pf2 = packW(P[i_pf_w2], 64, 128, 0, 128, 5);
  const unsigned int* pk_pf3 = packW(P[i_pf_w3], 128, 64, 0, 64, 5);
  const unsigned int* pk_qf1 = packW(P[i_qf_w1], 128, 64, 0, 64, 5);
  const unsigned int* pk_qf2 = packW(P[i_qf_w2], 64, 128, 0, 128, 5);
  const unsigned int* pk_qf3 = packW(P[i_qf_w3], 128, 64, 0, 64, 5);
  const unsigned int* pk_ic0 = packW(P[i_ic_wx], 256, 704, 0, 256, 5);   // repq
  const unsigned int* pk_ic1 = packW(P[i_ic_wx], 256, 704, 256, 256, 5); // repk
  const unsigned int* pk_ic2 = packW(P[i_ic_wx], 256, 704, 512, 64, 5);  // hiddg
  const unsigned int* pk_ic3 = packW(P[i_ic_wx], 256, 704, 576, 128, 5); // u
  const unsigned int* pk_ich = packW(P[i_ic_wh], 256, 64, 0, 64, 5);
  const unsigned int* pk_gc0 = packW(P[i_gc_wx], 256, 327, 0, 7, 5);     // vqb
  const unsigned int* pk_gc1 = packW(P[i_gc_wx], 256, 327, 7, 256, 5);   // repk
  const unsigned int* pk_gc2 = packW(P[i_gc_wx], 256, 327, 263, 64, 5);  // z
  const unsigned int* pk_gch = packW(P[i_gc_wh], 256, 64, 0, 64, 5);
  const unsigned int* pk_gd  = packW(P[i_gd_w], 128, 64, 0, 64, 5);
  const unsigned int* pk_go1 = packW(P[i_go_w1], 128, 128, 0, 128, 5);
  const unsigned int* pk_go2 = packW(P[i_go_w2], 64, 128, 0, 128, 5);
  const unsigned int* pk_go3 = packW(P[i_go_w3], 3, 64, 0, 64, 1);
  const unsigned int* pk_l0x = packW(P[i_l0_wih], 1024, 512, 0, 512, 1);
  const unsigned int* pk_l0h = packW(P[i_l0_whh], 1024, 256, 0, 256, 1);
  const unsigned int* pk_l1x = packW(P[i_l1_wih], 1024, 256, 0, 256, 1);
  const unsigned int* pk_l1h = packW(P[i_l1_whh], 1024, 256, 0, 256, 1);

  // ---- activation buffers ----
  unsigned short* enc_in = allocH(64LL * 10 * 4096);  // bf16 conv inputs
  unsigned short* tA     = allocH(64LL * 64 * 1024);
  unsigned short* tB     = allocH(64LL * 64 * 256);
  unsigned short* repq   = allocH(64LL * 256 * 256);
  unsigned short* repk   = allocH(64LL * 256 * 256);
  unsigned short* vqb    = allocH(64LL * 7 * 256);
  unsigned short* hiddg  = allocH(64LL * 64 * 256);
  unsigned short* hiddi  = allocH(64LL * 64 * 256);
  unsigned short* zbuf   = allocH(64LL * 64 * 256);
  unsigned short* ubufB  = allocH(64LL * 128 * 256);
  unsigned short* t128   = allocH(64LL * 128 * 256);
  unsigned short* t64    = allocH(64LL * 64 * 256);
  unsigned short* xTx    = allocH(512 * 16);
  unsigned short* xTh    = allocH(256 * 16);
  float* se3o  = allocF(60LL * 256 * 64);
  float* vecS  = allocF(20LL * 256);
  float* vecT  = allocF(60LL * 256);     // xtr pre-LSTM (4,15,256)
  float* xsr   = allocF(1024);
  float* xsrf  = allocF(1024);
  float* li    = allocF(4LL * 15 * 512);
  float* ys1   = allocF(4LL * 15 * 256);
  float* ys2   = allocF(4LL * 15 * 256);
  float* hL    = allocF(1024);
  float* cL    = allocF(1024);
  float* gT    = allocF(1024 * 16);
  float* cellg = allocF(64LL * 64 * 256);
  float* celli = allocF(64LL * 64 * 256);
  float* ubuf  = allocF(64LL * 128 * 256);
  float* gbuf  = allocF(64LL * 256 * 256);
  float* tfac  = allocF(64LL * 128 * 256);

  float* outX    = (float*)d_out;                 // (64,3,16,16)
  float* outQloc = outX + 64LL * 3 * 256;         // (16,64,64,16,16)
  float* outQlva = outQloc + 16LL * 64 * 64 * 256;
  float* outPloc = outQlva + 16LL * 64 * 64 * 256;
  float* outPlva = outPloc + 16LL * 64 * 64 * 256;

  // ---- scene encoder: context (K=5) ----
  k_concat_imgv<<<g256(20LL * 10 * 4096), 256, 0, stream>>>(xsk, vsk, enc_in, 20, 4096);
  conv(enc_in, pk_se1, P[i_se_b1], nullptr, tA, 20, 10, 64, 64, 64, 32, 32, 5, 2, 1, 1, 0);
  conv(tA, pk_se2, P[i_se_b2], nullptr, tB, 20, 64, 32, 32, 64, 16, 16, 5, 2, 1, 1, 0);
  conv(tB, pk_se3, P[i_se_b3], se3o, nullptr, 20, 64, 16, 16, 256, 8, 8, 5, 2, 1, 0, 0);
  k_mean_hw<<<g256(20 * 256), 256, 0, stream>>>(se3o, vecS, 20 * 256, 64, 1.f / 64.f);
  k_sum_ctx<<<g256(4 * 256), 256, 0, stream>>>(vecS, xsr, Bb, Kc, 256);

  // ---- scene encoder: targets (T=15) ----
  k_concat_imgv<<<g256(60LL * 10 * 4096), 256, 0, stream>>>(xtk, vtk, enc_in, 60, 4096);
  conv(enc_in, pk_se1, P[i_se_b1], nullptr, tA, 60, 10, 64, 64, 64, 32, 32, 5, 2, 1, 1, 0);
  conv(tA, pk_se2, P[i_se_b2], nullptr, tB, 60, 64, 32, 32, 64, 16, 16, 5, 2, 1, 1, 0);
  conv(tB, pk_se3, P[i_se_b3], se3o, nullptr, 60, 64, 16, 16, 256, 8, 8, 5, 2, 1, 0, 0);
  k_mean_hw<<<g256(60 * 256), 256, 0, stream>>>(se3o, vecT, 60 * 256, 64, 1.f / 64.f);
  k_addfr0<<<g256(1024), 256, 0, stream>>>(xsr, vecT, xsrf, Bb, 256, Tt);

  // ---- two-layer LSTM over T (WMMA GEMM, batch padded 4->16 cols) ----
  k_build_li<<<g256(4LL * 15 * 512), 256, 0, stream>>>(vecT, xsr, li, Bb, Tt, 256);
  fill0(hL, 1024); fill0(cL, 1024);
  for (int t = 0; t < Tt; ++t) {
    k_gatherT<<<g256(512 * 16), 256, 0, stream>>>(li, xTx, 512, Tt * 512, t * 512);
    k_gatherT<<<g256(256 * 16), 256, 0, stream>>>(hL, xTh, 256, 256, 0);
    conv(xTx, pk_l0x, P[i_l0_bih], gT, nullptr, 1, 512, 1, 16, 1024, 1, 16, 1, 1, 0, 0, 0);
    conv(xTh, pk_l0h, P[i_l0_bhh], gT, nullptr, 1, 256, 1, 16, 1024, 1, 16, 1, 1, 0, 0, 1);
    k_lstm_gate<<<g256(1024), 256, 0, stream>>>(gT, hL, cL, ys1, t, Tt);
  }
  fill0(hL, 1024); fill0(cL, 1024);
  for (int t = 0; t < Tt; ++t) {
    k_gatherT<<<g256(256 * 16), 256, 0, stream>>>(ys1, xTx, 256, Tt * 256, t * 256);
    k_gatherT<<<g256(256 * 16), 256, 0, stream>>>(hL, xTh, 256, 256, 0);
    conv(xTx, pk_l1x, P[i_l1_bih], gT, nullptr, 1, 256, 1, 16, 1024, 1, 16, 1, 1, 0, 0, 0);
    conv(xTh, pk_l1h, P[i_l1_bhh], gT, nullptr, 1, 256, 1, 16, 1024, 1, 16, 1, 1, 0, 0, 1);
    k_lstm_gate<<<g256(1024), 256, 0, stream>>>(gT, hL, cL, ys2, t, Tt);
  }

  // ---- inlet (query images) ----
  k_concat_imgv<<<g256(64LL * 10 * 4096), 256, 0, stream>>>(xq, vq, enc_in, NQ, 4096);
  conv(enc_in, pk_in1, P[i_in_b1], nullptr, tA, NQ, 10, 64, 64, 64, 32, 32, 5, 2, 1, 1, 0);
  conv(tA, pk_in2, P[i_in_b2], nullptr, repq, NQ, 64, 32, 32, 256, 16, 16, 5, 2, 1, 0, 0);

  // ---- loop-invariant broadcasts ----
  k_build_repk<<<g256(64LL * 256 * 256), 256, 0, stream>>>(xsrf, ys2, repk);
  k_bcast_v<<<g256(64LL * 7 * 256), 256, 0, stream>>>(vq, vqb, NQ, 7, 256);

  // ---- autoregressive scan ----
  fill0h(hiddg, 64LL * 64 * 256); fill0h(hiddi, 64LL * 64 * 256);
  fill0(cellg, 64LL * 64 * 256);  fill0(celli, 64LL * 64 * 256);
  fill0(ubuf, 64LL * 128 * 256);  fill0h(ubufB, 64LL * 128 * 256);
  for (int s = 0; s < AR; ++s) {
    long long so = (long long)s * 64 * 64 * 256;
    // prior factor on hiddg
    conv(hiddg, pk_pf1, P[i_pf_b1], nullptr, t128, NQ, 64, 16, 16, 128, 16, 16, 5, 1, 2, 1, 0);
    conv(t128, pk_pf2, P[i_pf_b2], nullptr, t64, NQ, 128, 16, 16, 64, 16, 16, 5, 1, 2, 1, 0);
    conv(t64, pk_pf3, P[i_pf_b3], tfac, nullptr, NQ, 64, 16, 16, 128, 16, 16, 5, 1, 2, 0, 0);
    k_split_fac<<<g256(64LL * 64 * 256), 256, 0, stream>>>(tfac, outPloc + so, outPlva + so, NQ, 64, 256);
    // inference ConvLSTM: segmented conv over concat(repq,repk,hiddg,u) + wh*hiddi
    conv(repq, pk_ic0, P[i_ic_b], gbuf, nullptr, NQ, 256, 16, 16, 256, 16, 16, 5, 1, 2, 0, 0);
    conv(repk, pk_ic1, nullptr, gbuf, nullptr, NQ, 256, 16, 16, 256, 16, 16, 5, 1, 2, 0, 1);
    conv(hiddg, pk_ic2, nullptr, gbuf, nullptr, NQ, 64, 16, 16, 256, 16, 16, 5, 1, 2, 0, 1);
    conv(ubufB, pk_ic3, nullptr, gbuf, nullptr, NQ, 128, 16, 16, 256, 16, 16, 5, 1, 2, 0, 1);
    conv(hiddi, pk_ich, nullptr, gbuf, nullptr, NQ, 64, 16, 16, 256, 16, 16, 5, 1, 2, 0, 1);
    k_convlstm_gate<<<g256(64LL * 64 * 256), 256, 0, stream>>>(gbuf, hiddi, celli, NQ, 64, 256);
    // posterior factor on hiddi
    conv(hiddi, pk_qf1, P[i_qf_b1], nullptr, t128, NQ, 64, 16, 16, 128, 16, 16, 5, 1, 2, 1, 0);
    conv(t128, pk_qf2, P[i_qf_b2], nullptr, t64, NQ, 128, 16, 16, 64, 16, 16, 5, 1, 2, 1, 0);
    conv(t64, pk_qf3, P[i_qf_b3], tfac, nullptr, NQ, 64, 16, 16, 128, 16, 16, 5, 1, 2, 0, 0);
    k_split_fac<<<g256(64LL * 64 * 256), 256, 0, stream>>>(tfac, outQloc + so, outQlva + so, NQ, 64, 256);
    k_z<<<g256(64LL * 64 * 256), 256, 0, stream>>>(tfac, eps + so, zbuf, NQ, 64, 256);
    // generator ConvLSTM: concat(vqb,repk,z) + wh*hiddg
    conv(vqb, pk_gc0, P[i_gc_b], gbuf, nullptr, NQ, 7, 16, 16, 256, 16, 16, 5, 1, 2, 0, 0);
    conv(repk, pk_gc1, nullptr, gbuf, nullptr, NQ, 256, 16, 16, 256, 16, 16, 5, 1, 2, 0, 1);
    conv(zbuf, pk_gc2, nullptr, gbuf, nullptr, NQ, 64, 16, 16, 256, 16, 16, 5, 1, 2, 0, 1);
    conv(hiddg, pk_gch, nullptr, gbuf, nullptr, NQ, 64, 16, 16, 256, 16, 16, 5, 1, 2, 0, 1);
    k_convlstm_gate<<<g256(64LL * 64 * 256), 256, 0, stream>>>(gbuf, hiddg, cellg, NQ, 64, 256);
    // u += conv(hiddg, gd) + b  (f32 master + bf16 mirror)
    conv(hiddg, pk_gd, P[i_gd_b], ubuf, ubufB, NQ, 64, 16, 16, 128, 16, 16, 5, 1, 2, 0, 1);
  }

  // ---- output decoder ----
  conv(ubufB, pk_go1, P[i_go_b1], nullptr, t128, NQ, 128, 16, 16, 128, 16, 16, 5, 1, 2, 1, 0);
  conv(t128, pk_go2, P[i_go_b2], nullptr, t64, NQ, 128, 16, 16, 64, 16, 16, 5, 1, 2, 1, 0);
  conv(t64, pk_go3, P[i_go_b3], outX, nullptr, NQ, 64, 16, 16, 3, 16, 16, 1, 1, 0, 0, 0);
}
